// SmallSwitchTransformer_45844480917647
// MI455X (gfx1250) — matrix operand order repaired and checked by reading, not compiled, not run on hardware
//
#include <hip/hip_runtime.h>
#include <hip/hip_bf16.h>
#include <stdint.h>

// ---- problem dims (compile-time) ----
#define Bb   8
#define Ss   512
#define Hh   512
#define NHh  8
#define DHh  64
#define Ee   8
#define Ll   6
#define Vv   32000
#define FFf  2048
#define TKN  (Bb*Ss)          // 4096 tokens

typedef __attribute__((ext_vector_type(16))) __bf16 v16bf;
typedef __attribute__((ext_vector_type(8)))  float  v8f;
typedef __attribute__((ext_vector_type(4)))  int    v4i;
typedef unsigned short u16;   // raw bf16 storage
typedef long long i64;

// gfx1250 async global->LDS path (ASYNCcnt), with safe fallback if the
// builtins are absent on this toolchain (host pass always takes fallback).
// Probe-verified param type: v4i* (generic), order (global src, lds dst).
#if __has_builtin(__builtin_amdgcn_global_load_async_to_lds_b128) && \
    __has_builtin(__builtin_amdgcn_s_wait_asynccnt)
#define USE_ASYNC_LDS 1
#define V4IP(x) ((v4i*)(void*)(x))
#else
#define USE_ASYNC_LDS 0
#endif

__device__ __forceinline__ u16 f2bf(float f) {
  union { float f; unsigned u; } x; x.f = f;
  unsigned r = x.u + 0x7FFFu + ((x.u >> 16) & 1u);   // round-to-nearest-even
  return (u16)(r >> 16);
}

enum { EPI_BF16_BIAS, EPI_F32_SCALE, EPI_BF16_PLAIN, EPI_F32_RESID_BIAS,
       EPI_BF16_RELU_BIAS, EPI_F32_RESID_GATE, EPI_F32_PLAIN };

// ============================================================================
// Generic WMMA GEMM:  C[M,N] = epilogue( A[M,K](bf16) * B(bf16) )
//  - B_KN=false: B is [N,K] row-major (weights);  B_KN=true: B is [K,N].
//  - GATHER: A rows (and C rows) indirected through perm[exOff[0]+m]; M
//    clamped to device count exCnt[0] (MoE routing without host sync).
//  - Batched via blockIdx.z: ptr offsets (z/zdiv)*X0 + (z%zdiv)*X1.
//  - Block: 256 thr = 8 waves; tile 128x128, K-step 32; wave = 32x64 =
//    2x4 tiles of v_wmma_f32_16x16x32_bf16.
//  - Double-buffered LDS; tile staging via GLOBAL_LOAD_ASYNC_TO_LDS_B128
//    when available (exact per-wave ASYNCcnt bookkeeping: OOB rows clamp
//    their address instead of dropping lanes, stores stay guarded).
// ============================================================================
template<bool B_KN, bool GATHER, int EPI>
__global__ __launch_bounds__(256) void k_gemm(
    const u16* __restrict__ A, const u16* __restrict__ Bm,
    const float* __restrict__ bias, float* __restrict__ Cf, u16* __restrict__ Cb,
    const float* __restrict__ resid, const float* __restrict__ gate,
    int M, int N, int K, int lda, int ldb, int ldc,
    int zdiv, i64 a0, i64 a1, i64 b0, i64 b1, i64 c0, i64 c1,
    float alpha,
    const int* __restrict__ perm, const int* __restrict__ exOff,
    const int* __restrict__ exCnt)
{
  const int z = blockIdx.z;
  A  += (i64)(z / zdiv) * a0 + (i64)(z % zdiv) * a1;
  Bm += (i64)(z / zdiv) * b0 + (i64)(z % zdiv) * b1;
  const i64 offC = (i64)(z / zdiv) * c0 + (i64)(z % zdiv) * c1;

  const int Meff = exCnt ? exCnt[0] : M;
  const int rowBase = blockIdx.y * 128;
  const int colBase = blockIdx.x * 128;
  if (rowBase >= Meff) return;

  __shared__ u16 As[2][128][32];
  __shared__ u16 Bs[2][128][32];

  const int tid  = threadIdx.x;
  const int lane = tid & 31, wave = tid >> 5;
  const int wm = (wave & 3) * 32;   // 4 waves along M
  const int wn = (wave >> 2) * 64;  // 2 waves along N

  v8f acc[2][4] = {};

  // per-stage async-instruction count per wave (constant across stages)
  constexpr int NASY = B_KN ? 1 : 2;

  // ---- one staging step: tile at K-offset kb into buffer `buf` ----
  auto stage = [&](int kb, int buf) {
    { // A tile: 128 rows x 32 K, one 16B chunk per thread
      int r = tid >> 1, hf = (tid & 1) * 16;
      int gr = rowBase + r;
      int grc = gr < Meff ? gr : 0;                       // clamp, keep lane live
      int arow = GATHER ? perm[exOff[0] + grc] : grc;
      const u16* src = A + (i64)arow * lda + kb + hf;
#if USE_ASYNC_LDS
      __builtin_amdgcn_global_load_async_to_lds_b128(V4IP(src), V4IP(&As[buf][r][hf]), 0, 0);
#else
      uint4 val = make_uint4(0u, 0u, 0u, 0u);
      if (gr < Meff) {
        val = *(const uint4*)src;
        if (kb + 32 < K) __builtin_prefetch((const void*)(src + 32), 0, 1);
      }
      *(uint4*)&As[buf][r][hf] = val;
#endif
    }
    if constexpr (!B_KN) { // B tile [N,K]: vectorized rows
      int r = tid >> 1, hf = (tid & 1) * 16;
      int gn = colBase + r;
      int gnc = gn < N ? gn : 0;
      const u16* src = Bm + (i64)gnc * ldb + kb + hf;
#if USE_ASYNC_LDS
      __builtin_amdgcn_global_load_async_to_lds_b128(V4IP(src), V4IP(&Bs[buf][r][hf]), 0, 0);
#else
      uint4 val = make_uint4(0u, 0u, 0u, 0u);
      if (gn < N) {
        val = *(const uint4*)src;
        if (kb + 32 < K) __builtin_prefetch((const void*)(src + 32), 0, 1);
      }
      *(uint4*)&Bs[buf][r][hf] = val;
#endif
    } else {               // B tile [K,N]: transpose into LDS (scalar)
      for (int i2 = tid; i2 < 128 * 32; i2 += 256) {
        int n = i2 >> 5, k = i2 & 31;
        int gn = colBase + n;
        u16 vv = 0;
        if (gn < N) vv = Bm[(i64)(kb + k) * ldb + gn];
        Bs[buf][n][k] = vv;
      }
    }
  };

  const int NT = K >> 5;   // K multiple of 32 for every call site
  stage(0, 0);

  for (int it = 0; it < NT; ++it) {
    const int buf = it & 1;
    if (it + 1 < NT) {
      stage((it + 1) << 5, buf ^ 1);
#if USE_ASYNC_LDS
      __builtin_amdgcn_s_wait_asynccnt(NASY);   // retire previous stage only
#endif
    } else {
#if USE_ASYNC_LDS
      __builtin_amdgcn_s_wait_asynccnt(0);
#endif
    }
    __syncthreads();

    // ---- fragments per ISA 7.12.2 layouts ----
    const int r16 = lane & 15, hi = lane >> 4;
    v16bf afr[2], bfr[4];
#pragma unroll
    for (int i = 0; i < 2; i++) {
      union { uint4 q[2]; v16bf v; } u;
      // A 16x32 bf16: lane half hi -> K chunks [hi*8, hi*8+8) and [hi*8+16, hi*8+24)
      u.q[0] = *(const uint4*)&As[buf][wm + i * 16 + r16][hi * 8];
      u.q[1] = *(const uint4*)&As[buf][wm + i * 16 + r16][hi * 8 + 16];
      afr[i] = u.v;
    }
#pragma unroll
    for (int j = 0; j < 4; j++) {
      union { uint4 q[2]; v16bf v; } u;
      // B 32x16 bf16: lane = column, K chunks [hi*8, hi*8+8) and [hi*8+16, hi*8+24)
      u.q[0] = *(const uint4*)&Bs[buf][wn + j * 16 + r16][hi * 8];
      u.q[1] = *(const uint4*)&Bs[buf][wn + j * 16 + r16][hi * 8 + 16];
      bfr[j] = u.v;
    }
#pragma unroll
    for (int i = 0; i < 2; i++)
#pragma unroll
      for (int j = 0; j < 4; j++)
        acc[i][j] = __builtin_amdgcn_wmma_f32_16x16x32_bf16(
            false, afr[i], false, bfr[j], (short)0, acc[i][j], false, false);
    __syncthreads();
  }

  // ---- epilogue: C/D layout row = 8*(lane>>4)+v, col = lane&15 ----
  const int r16 = lane & 15, hi8 = (lane >> 4) * 8;
#pragma unroll
  for (int i = 0; i < 2; i++) {
#pragma unroll
    for (int j = 0; j < 4; j++) {
      int gn = colBase + wn + j * 16 + r16;
      if (gn >= N) continue;
      float bsv = 0.f;
      if constexpr (EPI == EPI_BF16_BIAS || EPI == EPI_F32_RESID_BIAS ||
                    EPI == EPI_BF16_RELU_BIAS || EPI == EPI_F32_RESID_GATE)
        bsv = bias[gn];
#pragma unroll
      for (int vv = 0; vv < 8; vv++) {
        int gm = rowBase + wm + i * 16 + hi8 + vv;
        if (gm >= Meff) continue;
        i64 crow = GATHER ? (i64)perm[exOff[0] + gm] : (i64)gm;
        float val = acc[i][j][vv];
        if constexpr (EPI == EPI_BF16_BIAS)
          Cb[offC + crow * ldc + gn] = f2bf(val + bsv);
        else if constexpr (EPI == EPI_F32_SCALE)
          Cf[offC + crow * ldc + gn] = val * alpha;
        else if constexpr (EPI == EPI_BF16_PLAIN)
          Cb[offC + crow * ldc + gn] = f2bf(val);
        else if constexpr (EPI == EPI_F32_RESID_BIAS)
          Cf[offC + crow * ldc + gn] = resid[crow * ldc + gn] + val + bsv;
        else if constexpr (EPI == EPI_BF16_RELU_BIAS) {
          float r = val + bsv;
          Cb[offC + crow * ldc + gn] = f2bf(r > 0.f ? r : 0.f);
        } else if constexpr (EPI == EPI_F32_RESID_GATE)
          Cf[offC + crow * ldc + gn] =
              resid[crow * ldc + gn] + gate[crow] * (val + bsv);
        else
          Cf[offC + crow * ldc + gn] = val;
      }
    }
  }
}

// ============================================================================
// Small kernels
// ============================================================================
__global__ __launch_bounds__(256) void k_embed(const int* __restrict__ ids,
    const float* __restrict__ emb, const float* __restrict__ pos,
    float* __restrict__ xf, u16* __restrict__ xb)
{
  i64 i = (i64)blockIdx.x * 256 + threadIdx.x;   // grid exactly T*H/256
  int t = (int)(i >> 9), h = (int)(i & 511);
  float v = emb[(i64)ids[t] * Hh + h] + pos[(i64)(t & (Ss - 1)) * Hh + h];
  xf[i] = v; xb[i] = f2bf(v);
}

__global__ __launch_bounds__(256) void k_cvt(const float* __restrict__ s,
                                             u16* __restrict__ d, i64 n)
{
  i64 i = (i64)blockIdx.x * 256 + threadIdx.x;
  if (i < n) d[i] = f2bf(s[i]);
}

__global__ __launch_bounds__(256) void k_softmax512(const float* __restrict__ S,
                                                    u16* __restrict__ P)
{
  const i64 row = blockIdx.x;
  const float* src = S + row * Ss;
  u16* dst = P + row * Ss;
  int tid = threadIdx.x;
  float a = src[tid], b = src[tid + 256];
  __shared__ float red[256];
  red[tid] = fmaxf(a, b); __syncthreads();
  for (int s = 128; s > 0; s >>= 1) { if (tid < s) red[tid] = fmaxf(red[tid], red[tid + s]); __syncthreads(); }
  float mx = red[0]; __syncthreads();
  float ea = __expf(a - mx), eb = __expf(b - mx);
  red[tid] = ea + eb; __syncthreads();
  for (int s = 128; s > 0; s >>= 1) { if (tid < s) red[tid] += red[tid + s]; __syncthreads(); }
  float inv = 1.f / red[0];
  dst[tid] = f2bf(ea * inv); dst[tid + 256] = f2bf(eb * inv);
}

__global__ __launch_bounds__(256) void k_layernorm(const float* __restrict__ in,
    const float* __restrict__ g, const float* __restrict__ bt,
    float* __restrict__ outf, u16* __restrict__ outb)
{
  const i64 row = blockIdx.x;
  const float* src = in + row * Hh;
  int tid = threadIdx.x;
  float a = src[tid], b = src[tid + 256];
  __shared__ float red[256];
  red[tid] = a + b; __syncthreads();
  for (int s = 128; s > 0; s >>= 1) { if (tid < s) red[tid] += red[tid + s]; __syncthreads(); }
  float mean = red[0] * (1.f / (float)Hh); __syncthreads();
  float da = a - mean, db = b - mean;
  red[tid] = da * da + db * db; __syncthreads();
  for (int s = 128; s > 0; s >>= 1) { if (tid < s) red[tid] += red[tid + s]; __syncthreads(); }
  float rstd = rsqrtf(red[0] * (1.f / (float)Hh) + 1e-5f);
  float ya = da * rstd * g[tid] + bt[tid];
  float yb = db * rstd * g[tid + 256] + bt[tid + 256];
  outf[row * Hh + tid] = ya;       outf[row * Hh + tid + 256] = yb;
  outb[row * Hh + tid] = f2bf(ya); outb[row * Hh + tid + 256] = f2bf(yb);
}

// top-1 switch gate: one wave per token
__global__ __launch_bounds__(256) void k_gate(const float* __restrict__ x,
    const float* __restrict__ gw, float* __restrict__ probs,
    int* __restrict__ expertOf, float* __restrict__ gateVal,
    int* __restrict__ counts)
{
  int token = blockIdx.x * 8 + (threadIdx.x >> 5);
  int lane  = threadIdx.x & 31;
  const float* xr = x + (i64)token * Hh;
  float s[Ee];
#pragma unroll
  for (int e = 0; e < Ee; e++) {
    const float* w = gw + (i64)e * Hh;
    float acc = 0.f;
    for (int k = lane; k < Hh; k += 32) acc += xr[k] * w[k];
#pragma unroll
    for (int off = 16; off > 0; off >>= 1) acc += __shfl_xor(acc, off, 32);
    s[e] = acc;
  }
  float mx = s[0];
#pragma unroll
  for (int e = 1; e < Ee; e++) mx = fmaxf(mx, s[e]);
  float den = 0.f;
#pragma unroll
  for (int e = 0; e < Ee; e++) { s[e] = __expf(s[e] - mx); den += s[e]; }
  float inv = 1.f / den;
#pragma unroll
  for (int e = 0; e < Ee; e++) s[e] *= inv;
  int best = 0; float bv = s[0];
#pragma unroll
  for (int e = 1; e < Ee; e++) if (s[e] > bv) { bv = s[e]; best = e; }
  if (lane == 0) {
    expertOf[token] = best;
    gateVal[token]  = bv / (bv + 1e-8f);
    atomicAdd(&counts[best], 1);
#pragma unroll
    for (int e = 0; e < Ee; e++) probs[(i64)token * Ee + e] = s[e];
  }
}

// deterministic load-balance loss per layer (fixed tree reduction)
__global__ __launch_bounds__(256) void k_lb(const float* __restrict__ probs,
    const int* __restrict__ expertOf, float* __restrict__ lbs, int layer)
{
  int tid = threadIdx.x;
  float rp[Ee] = {}, eu[Ee] = {};
  for (int t = tid; t < TKN; t += 256) {
    eu[expertOf[t]] += 1.f;
#pragma unroll
    for (int e = 0; e < Ee; e++) rp[e] += probs[(i64)t * Ee + e];
  }
  __shared__ float red[256];
  float lb = 0.f;
  for (int e = 0; e < Ee; e++) {
    red[tid] = rp[e]; __syncthreads();
    for (int s = 128; s > 0; s >>= 1) { if (tid < s) red[tid] += red[tid + s]; __syncthreads(); }
    float rps = red[0]; __syncthreads();
    red[tid] = eu[e]; __syncthreads();
    for (int s = 128; s > 0; s >>= 1) { if (tid < s) red[tid] += red[tid + s]; __syncthreads(); }
    float eus = red[0]; __syncthreads();
    lb += (rps / (float)TKN) * (eus / (float)TKN);
  }
  if (tid == 0) lbs[layer] = (float)Ee * lb;
}

__global__ void k_scan(const int* __restrict__ counts, int* __restrict__ offs,
                       int* __restrict__ cursor)
{
  if (threadIdx.x == 0) {
    int a = 0;
    for (int e = 0; e < Ee; e++) { offs[e] = a; cursor[e] = a; a += counts[e]; }
    offs[Ee] = a;
  }
}

__global__ __launch_bounds__(256) void k_scatter(const int* __restrict__ expertOf,
    int* __restrict__ cursor, int* __restrict__ perm)
{
  int t = blockIdx.x * 256 + threadIdx.x;
  if (t < TKN) {
    int e = expertOf[t];
    int p = atomicAdd(&cursor[e], 1);
    perm[p] = t;
  }
}

__global__ void k_loss(const float* __restrict__ lbs, float* __restrict__ o)
{
  if (threadIdx.x == 0 && blockIdx.x == 0) {
    float s = 0.f;
    for (int l = 0; l < Ll; l++) s += lbs[l];
    o[0] = s / (float)Ll;
  }
}

// ============================================================================
// Host orchestration
// ============================================================================
extern "C" void kernel_launch(void* const* d_in, const int* in_sizes, int n_in,
                              void* d_out, int out_size, void* d_ws, size_t ws_size,
                              hipStream_t stream)
{
  (void)in_sizes; (void)n_in; (void)out_size; (void)ws_size;
  const int*   ids        = (const int*)  d_in[0];
  const float* emb        = (const float*)d_in[1];
  const float* pos_emb    = (const float*)d_in[2];
  const float* attn_in_w  = (const float*)d_in[3];
  const float* attn_in_b  = (const float*)d_in[4];
  const float* attn_out_w = (const float*)d_in[5];
  const float* attn_out_b = (const float*)d_in[6];
  const float* ln1_g      = (const float*)d_in[7];
  const float* ln1_b      = (const float*)d_in[8];
  const float* gate_w     = (const float*)d_in[9];
  const float* w1         = (const float*)d_in[10];
  const float* b1         = (const float*)d_in[11];
  const float* w2         = (const float*)d_in[12];
  const float* b2         = (const float*)d_in[13];
  const float* ln2_g      = (const float*)d_in[14];
  const float* ln2_b      = (const float*)d_in[15];
  const float* lnf_g      = (const float*)d_in[16];
  const float* lnf_b      = (const float*)d_in[17];
  const float* lm_w       = (const float*)d_in[18];
  float* out = (float*)d_out;

  // ---- carve workspace ----
  char* p = (char*)d_ws;
  auto carve = [&](size_t bytes) -> void* {
    void* r = (void*)p; p += (bytes + 255) & ~(size_t)255; return r;
  };
  u16*   wb_qkvw = (u16*)  carve((size_t)Ll * 3 * Hh * Hh * 2);   // 4.72M
  u16*   wb_outw = (u16*)  carve((size_t)Ll * Hh * Hh * 2);       // 1.57M
  u16*   wb_w1   = (u16*)  carve((size_t)Ll * Ee * FFf * Hh * 2); // 50.3M
  u16*   wb_w2   = (u16*)  carve((size_t)Ll * Ee * Hh * FFf * 2); // 50.3M
  u16*   wb_lm   = (u16*)  carve((size_t)Vv * Hh * 2);            // 16.4M
  float* xf      = (float*)carve((size_t)TKN * Hh * 4);
  float* tmp     = (float*)carve((size_t)TKN * Hh * 4);
  u16*   xb      = (u16*)  carve((size_t)TKN * Hh * 2);
  u16*   qkvb    = (u16*)  carve((size_t)TKN * 3 * Hh * 2);
  float* scores  = (float*)carve((size_t)Bb * NHh * Ss * Ss * 4); // 64MB
  u16*   pb      = (u16*)  carve((size_t)Bb * NHh * Ss * Ss * 2); // 32MB
  u16*   ob      = (u16*)  carve((size_t)TKN * Hh * 2);
  u16*   hidb    = (u16*)  carve((size_t)TKN * FFf * 2);          // 16MB
  float* probs   = (float*)carve((size_t)TKN * Ee * 4);
  float* gateVal = (float*)carve((size_t)TKN * 4);
  int*   expertOf= (int*)  carve((size_t)TKN * 4);
  int*   counts  = (int*)  carve(Ee * 4);
  int*   offs    = (int*)  carve((Ee + 1) * 4);
  int*   cursor  = (int*)  carve(Ee * 4);
  int*   perm    = (int*)  carve((size_t)TKN * 4);
  float* lbs     = (float*)carve(Ll * 4);

  const dim3 blk(256, 1, 1);
  auto G1 = [](i64 n) { return dim3((unsigned)((n + 255) / 256), 1, 1); };

  // ---- fp32 -> bf16 weight conversion (once per call) ----
  k_cvt<<<G1((i64)Ll * 3 * Hh * Hh), blk, 0, stream>>>(attn_in_w, wb_qkvw, (i64)Ll * 3 * Hh * Hh);
  k_cvt<<<G1((i64)Ll * Hh * Hh),     blk, 0, stream>>>(attn_out_w, wb_outw, (i64)Ll * Hh * Hh);
  k_cvt<<<G1((i64)Ll * Ee * FFf * Hh), blk, 0, stream>>>(w1, wb_w1, (i64)Ll * Ee * FFf * Hh);
  k_cvt<<<G1((i64)Ll * Ee * Hh * FFf), blk, 0, stream>>>(w2, wb_w2, (i64)Ll * Ee * Hh * FFf);
  k_cvt<<<G1((i64)Vv * Hh),          blk, 0, stream>>>(lm_w, wb_lm, (i64)Vv * Hh);

  // ---- embedding ----
  k_embed<<<G1((i64)TKN * Hh), blk, 0, stream>>>(ids, emb, pos_emb, xf, xb);

  for (int l = 0; l < Ll; l++) {
    // QKV: [T,3H] bf16 = xb @ in_w^T + b
    k_gemm<false, false, EPI_BF16_BIAS><<<dim3(12, 32, 1), blk, 0, stream>>>(
        xb, wb_qkvw + (i64)l * 3 * Hh * Hh, attn_in_b + (i64)l * 3 * Hh,
        nullptr, qkvb, nullptr, nullptr,
        TKN, 3 * Hh, Hh, Hh, Hh, 3 * Hh,
        1, 0, 0, 0, 0, 0, 0, 1.f, nullptr, nullptr, nullptr);

    // scores[b,h] = q @ k^T / 8   (z = b*NH+h)
    k_gemm<false, false, EPI_F32_SCALE><<<dim3(4, 4, Bb * NHh), blk, 0, stream>>>(
        qkvb, qkvb + Hh, nullptr, scores, nullptr, nullptr, nullptr,
        Ss, Ss, DHh, 3 * Hh, 3 * Hh, Ss,
        NHh, (i64)Ss * 3 * Hh, DHh, (i64)Ss * 3 * Hh, DHh,
        (i64)NHh * Ss * Ss, (i64)Ss * Ss,
        0.125f, nullptr, nullptr, nullptr);

    k_softmax512<<<dim3(Bb * NHh * Ss, 1, 1), blk, 0, stream>>>(scores, pb);

    // o[b,h] = p @ v   (B is K-major -> transpose in LDS), store bf16
    k_gemm<true, false, EPI_BF16_PLAIN><<<dim3(1, 4, Bb * NHh), blk, 0, stream>>>(
        pb, qkvb + 2 * Hh, nullptr, nullptr, ob, nullptr, nullptr,
        Ss, DHh, Ss, Ss, 3 * Hh, Hh,
        NHh, (i64)NHh * Ss * Ss, (i64)Ss * Ss, (i64)Ss * 3 * Hh, DHh,
        (i64)Ss * Hh, DHh,
        1.f, nullptr, nullptr, nullptr);

    // out-proj + residual: tmp = x + o @ out_w^T + b
    k_gemm<false, false, EPI_F32_RESID_BIAS><<<dim3(4, 32, 1), blk, 0, stream>>>(
        ob, wb_outw + (i64)l * Hh * Hh, attn_out_b + (i64)l * Hh,
        tmp, nullptr, xf, nullptr,
        TKN, Hh, Hh, Hh, Hh, Hh,
        1, 0, 0, 0, 0, 0, 0, 1.f, nullptr, nullptr, nullptr);

    k_layernorm<<<dim3(TKN, 1, 1), blk, 0, stream>>>(
        tmp, ln1_g + (i64)l * Hh, ln1_b + (i64)l * Hh, xf, xb);

    // ---- top-1 routing ----
    (void)hipMemsetAsync(counts, 0, Ee * sizeof(int), stream);
    k_gate<<<dim3(TKN / 8, 1, 1), blk, 0, stream>>>(
        xf, gate_w + (i64)l * Ee * Hh, probs, expertOf, gateVal, counts);
    k_lb<<<dim3(1, 1, 1), blk, 0, stream>>>(probs, expertOf, lbs, l);
    k_scan<<<dim3(1, 1, 1), dim3(32, 1, 1), 0, stream>>>(counts, offs, cursor);
    k_scatter<<<dim3(TKN / 256, 1, 1), blk, 0, stream>>>(expertOf, cursor, perm);

    // ---- per-expert routed FFN (grids sized worst-case, early-exit on count) --
    for (int e = 0; e < Ee; e++) {
      const u16*   w1e = wb_w1 + (i64)(l * Ee + e) * FFf * Hh;
      const u16*   w2e = wb_w2 + (i64)(l * Ee + e) * Hh * FFf;
      const float* b1e = b1 + (i64)(l * Ee + e) * FFf;
      const float* b2e = b2 + (i64)(l * Ee + e) * Hh;
      // hid[t] = relu(x[t] @ w1^T + b1), gathered rows
      k_gemm<false, true, EPI_BF16_RELU_BIAS><<<dim3(16, 32, 1), blk, 0, stream>>>(
          xb, w1e, b1e, nullptr, hidb, nullptr, nullptr,
          TKN, FFf, Hh, Hh, Hh, FFf,
          1, 0, 0, 0, 0, 0, 0, 1.f, perm, offs + e, counts + e);
      // tmp[t] = x[t] + gate[t]*(hid[t] @ w2^T + b2)   (disjoint token rows)
      k_gemm<false, true, EPI_F32_RESID_GATE><<<dim3(4, 32, 1), blk, 0, stream>>>(
          hidb, w2e, b2e, tmp, nullptr, xf, gateVal,
          TKN, Hh, FFf, FFf, FFf, Hh,
          1, 0, 0, 0, 0, 0, 0, 1.f, perm, offs + e, counts + e);
    }

    k_layernorm<<<dim3(TKN, 1, 1), blk, 0, stream>>>(
        tmp, ln2_g + (i64)l * Hh, ln2_b + (i64)l * Hh, xf, xb);
  }

  // ---- final LN + LM head ----
  k_layernorm<<<dim3(TKN, 1, 1), blk, 0, stream>>>(xf, lnf_g, lnf_b, tmp, xb);
  k_gemm<false, false, EPI_F32_PLAIN><<<dim3(250, 32, 1), blk, 0, stream>>>(
      xb, wb_lm, nullptr, out, nullptr, nullptr, nullptr,
      TKN, Vv, Hh, Hh, Hh, Vv,
      1, 0, 0, 0, 0, 0, 0, 1.f, nullptr, nullptr, nullptr);

  k_loss<<<dim3(1, 1, 1), dim3(32, 1, 1), 0, stream>>>(lbs, out + (i64)TKN * Vv);
}
